// MMDLoss_85014582657035
// MI455X (gfx1250) — matrix operand order repaired and checked by reading, not compiled, not run on hardware
//
#include <hip/hip_runtime.h>
#include <hip/hip_bf16.h>

typedef __attribute__((ext_vector_type(16))) _Float16 v16h;
typedef __attribute__((ext_vector_type(8)))  _Float16 v8h;
typedef __attribute__((ext_vector_type(8)))  float    v8f;

#define NROWS 8192      // rows in X (== rows in Y)
#define DIM   128       // feature dim
#define NT    16384     // total rows of Z = [X;Y]
#define BT    128       // output tile edge per block
#define NTILE (NT / BT) // 128 tiles per dimension

// ---------------------------------------------------------------------------
// Kernel 1: split Z rows into f16 hi/lo parts and compute row squared norms.
// One block (128 threads) per row; fully coalesced.
// ---------------------------------------------------------------------------
__global__ __launch_bounds__(DIM) void mmd_prep_kernel(
    const float* __restrict__ X, const float* __restrict__ Y,
    _Float16* __restrict__ zhi, _Float16* __restrict__ zlo,
    float* __restrict__ sq) {
  const int row = blockIdx.x;
  const int tid = threadIdx.x;
  const float* src = (row < NROWS) ? (X + (size_t)row * DIM)
                                   : (Y + (size_t)(row - NROWS) * DIM);
  float x = src[tid];
  _Float16 h = (_Float16)x;          // hi part (RNE)
  float r = x - (float)h;            // residual, |r| <~ 2^-11 |x|
  size_t idx = (size_t)row * DIM + tid;
  zhi[idx] = h;
  zlo[idx] = (_Float16)r;

  __shared__ float s[DIM];
  s[tid] = x * x;                    // norms from full f32 data (exact path)
  __syncthreads();
  #pragma unroll
  for (int off = DIM / 2; off > 0; off >>= 1) {
    if (tid < off) s[tid] += s[tid + off];
    __syncthreads();
  }
  if (tid == 0) sq[row] = s[0];
}

// ---------------------------------------------------------------------------
// Kernel 2: fused tiled Gram + RBF + signed reduction.
// Block = 256 threads = 8 waves; block tile = 128x128 of the virtual 16384^2
// kernel matrix. Wave w owns rows [R0+16w, R0+16w+16) x all 128 block cols.
// G is accumulated via split-f16 WMMA: hi*hi + hi*lo + lo*hi.
// ---------------------------------------------------------------------------
__global__ __launch_bounds__(256) void mmd_tile_kernel(
    const _Float16* __restrict__ zhi, const _Float16* __restrict__ zlo,
    const float* __restrict__ sq, float* __restrict__ partial) {
  const int lane = threadIdx.x & 31;
  const int wave = threadIdx.x >> 5;          // 0..7
  const int C0 = blockIdx.x * BT;             // column tile origin
  const int R0 = blockIdx.y * BT;             // row tile origin

  // A-fragment addressing (16x32 f16, ISA 7.12.2): lane holds row M=lane%16,
  // K = 0..15 (lanes 0-15) or 16..31 (lanes 16-31), contiguous: this code's order, not the ISA table's (matches B).
  const int arow  = R0 + wave * 16 + (lane & 15);
  const int koffA = (lane < 16) ? 0 : 16;
  // B-fragment addressing (32x16): lane holds col N=lane%16,
  // K = 0..15 (lanes 0-15) or 16..31 (lanes 16-31), contiguous.
  const int koffB = (lane < 16) ? 0 : 16;
  const int ncol0 = C0 + (lane & 15);

  v8f acc[8] = {};  // 8 col-tiles of 16x16 f32 accumulators

  #pragma unroll
  for (int kc = 0; kc < 4; ++kc) {            // K = 128 in chunks of 32
    const int k0 = kc * 32;
    const _Float16* pah = zhi + (size_t)arow * DIM + (k0 + koffA);
    const _Float16* pal = zlo + (size_t)arow * DIM + (k0 + koffA);
    v8h a0h = *(const v8h*)(pah);
    v8h a1h = *(const v8h*)(pah + 8);
    v8h a0l = *(const v8h*)(pal);
    v8h a1l = *(const v8h*)(pal + 8);
    v16h ahi = __builtin_shufflevector(a0h, a1h, 0,1,2,3,4,5,6,7,8,9,10,11,12,13,14,15);
    v16h alo = __builtin_shufflevector(a0l, a1l, 0,1,2,3,4,5,6,7,8,9,10,11,12,13,14,15);

    #pragma unroll
    for (int t = 0; t < 8; ++t) {
      const int col = ncol0 + t * 16;
      const size_t boff = (size_t)col * DIM + (k0 + koffB);
      v16h bhi = *(const v16h*)(zhi + boff);
      v16h blo = *(const v16h*)(zlo + boff);
      // G += Ahi*Bhi + Ahi*Blo + Alo*Bhi   (lo*lo term ~2^-22, dropped)
      acc[t] = __builtin_amdgcn_wmma_f32_16x16x32_f16(
          false, ahi, false, bhi, (short)0, acc[t], false, false);
      acc[t] = __builtin_amdgcn_wmma_f32_16x16x32_f16(
          false, ahi, false, blo, (short)0, acc[t], false, false);
      acc[t] = __builtin_amdgcn_wmma_f32_16x16x32_f16(
          false, alo, false, bhi, (short)0, acc[t], false, false);
    }
  }

  // Fused epilogue. C/D layout: VGPR j -> M = mbase + j, N = ncol0 (+16t).
  const int mbase = R0 + wave * 16 + ((lane < 16) ? 0 : 8);
  float sqm[8];
  #pragma unroll
  for (int j = 0; j < 8; ++j) sqm[j] = sq[mbase + j];
  const bool mIsX = (mbase < NROWS);  // mbase is a multiple of 8; all j same side

  float local = 0.0f;
  #pragma unroll
  for (int t = 0; t < 8; ++t) {
    const int n = ncol0 + t * 16;
    const float sqn = sq[n];
    const float sgn = (((n < NROWS)) == mIsX) ? 1.0f : -1.0f;
    #pragma unroll
    for (int j = 0; j < 8; ++j) {
      float d2 = sqm[j] + sqn - 2.0f * acc[t][j];
      d2 = fmaxf(d2, 0.0f);
      local += sgn * __expf(d2 * (-1.0f / (2.0f * 10.0f * 10.0f)));
    }
  }

  // Deterministic block reduction (no float atomics -> bitwise-stable output).
  __shared__ float red[256];
  red[threadIdx.x] = local;
  __syncthreads();
  #pragma unroll
  for (int off = 128; off > 0; off >>= 1) {
    if (threadIdx.x < off) red[threadIdx.x] += red[threadIdx.x + off];
    __syncthreads();
  }
  if (threadIdx.x == 0)
    partial[blockIdx.y * gridDim.x + blockIdx.x] = red[0];
}

// ---------------------------------------------------------------------------
// Kernel 3: reduce 16384 block partials (double accumulation) -> scalar.
// ---------------------------------------------------------------------------
__global__ __launch_bounds__(256) void mmd_reduce_kernel(
    const float* __restrict__ partial, float* __restrict__ out) {
  __shared__ double s[256];
  double acc = 0.0;
  for (int i = threadIdx.x; i < NTILE * NTILE; i += 256)
    acc += (double)partial[i];
  s[threadIdx.x] = acc;
  __syncthreads();
  #pragma unroll
  for (int off = 128; off > 0; off >>= 1) {
    if (threadIdx.x < off) s[threadIdx.x] += s[threadIdx.x + off];
    __syncthreads();
  }
  if (threadIdx.x == 0)
    out[0] = (float)(s[0] / ((double)NROWS * (double)NROWS));
}

// ---------------------------------------------------------------------------
extern "C" void kernel_launch(void* const* d_in, const int* in_sizes, int n_in,
                              void* d_out, int out_size, void* d_ws, size_t ws_size,
                              hipStream_t stream) {
  const float* X = (const float*)d_in[0];  // 8192*128 f32
  const float* Y = (const float*)d_in[1];  // 8192*128 f32
  float* out = (float*)d_out;              // 1 f32

  // Workspace layout (~8.2 MB total):
  //   [0, 4MB)        Zhi  f16[16384*128]
  //   [4MB, 8MB)      Zlo  f16[16384*128]
  //   [8MB, +64KB)    sq   f32[16384]
  //   [+64KB, +128KB) partial f32[128*128]
  char* ws = (char*)d_ws;
  _Float16* zhi   = (_Float16*)(ws);
  _Float16* zlo   = (_Float16*)(ws + (size_t)NT * DIM * sizeof(_Float16));
  float*    sq    = (float*)   (ws + (size_t)NT * DIM * sizeof(_Float16) * 2);
  float*    part  = (float*)   (ws + (size_t)NT * DIM * sizeof(_Float16) * 2
                                   + (size_t)NT * sizeof(float));

  mmd_prep_kernel<<<NT, DIM, 0, stream>>>(X, Y, zhi, zlo, sq);

  dim3 grid(NTILE, NTILE);
  mmd_tile_kernel<<<grid, 256, 0, stream>>>(zhi, zlo, sq, part);

  mmd_reduce_kernel<<<1, 256, 0, stream>>>(part, out);
}
